// ConvLayer_74577812128030
// MI455X (gfx1250) — matrix operand order repaired and checked by reading, not compiled, not run on hardware
//
#include <hip/hip_runtime.h>

typedef __attribute__((ext_vector_type(16))) _Float16 v16h;
typedef __attribute__((ext_vector_type(8)))  _Float16 v8h;
typedef __attribute__((ext_vector_type(8)))  float    v8f;

#define OCN    96
#define KSZ    147
#define KPAD   160
#define OH     112
#define OW     112
#define IHH    224
#define IWW    224
#define ICN    3
#define BATCH  16
#define PLANE  (OH*OW)          // 12544
#define NPLANES (BATCH*OCN)     // 1536
#define POOLN  (NPLANES*56*56)  // 4816896

// ---------------- weight renorm + f16 pack (N-major, K padded to 160) -------
__global__ void wrenorm_pack(const float* __restrict__ w, _Float16* __restrict__ wpack) {
    int t = threadIdx.x;
    if (t >= OCN) return;
    const float* wf = w + t * KSZ;
    float ss = 0.f;
    for (int k = 0; k < KSZ; ++k) ss += wf[k] * wf[k];
    float rms   = sqrtf(ss / (float)KSZ);
    float scale = (rms > 0.1f) ? (0.1f / rms) : 1.0f;
    _Float16* dst = wpack + t * KPAD;
    for (int k = 0; k < KSZ; ++k)  dst[k] = (_Float16)(wf[k] * scale);
    for (int k = KSZ; k < KPAD; ++k) dst[k] = (_Float16)0.f;
}

// ---------------- implicit-GEMM conv via v_wmma_f32_16x16x32_f16 ------------
__global__ __launch_bounds__(128) void conv_wmma(
    const float* __restrict__ x, const _Float16* __restrict__ wpack,
    const float* __restrict__ bias, float* __restrict__ y)
{
    __shared__ _Float16 lA[64 * KPAD];   // 20 KB
    const int tid   = threadIdx.x;
    const int wave  = tid >> 5;
    const int lane  = tid & 31;
    const int mbase = blockIdx.x * 64;

    for (int i = tid; i < 64 * KPAD; i += 128) {
        int row = i / KPAD, k = i % KPAD;
        float v = 0.f;
        if (k < KSZ) {
            int m  = mbase + row;
            int ow = m % OW; int tmp = m / OW; int oh = tmp % OH; int b = tmp / OH;
            int ic = k / 49; int r = k % 49; int kh = r / 7; int kw = r % 7;
            int ih = oh * 2 - 3 + kh, iw = ow * 2 - 3 + kw;
            if (ih >= 0 && ih < IHH && iw >= 0 && iw < IWW)
                v = x[((b * ICN + ic) * IHH + ih) * IWW + iw];
        }
        lA[i] = (_Float16)v;
    }
    __syncthreads();

    v8f acc[6];
    #pragma unroll
    for (int nt = 0; nt < 6; ++nt) { v8f z = {}; acc[nt] = z; }

    const int arow   = wave * 16 + (lane & 15);
    const int abase0 = (lane >> 4) * 16;
    const int bn     = lane & 15;
    const int bkoff  = (lane >> 4) * 16;

    #pragma unroll
    for (int ks = 0; ks < 5; ++ks) {
        const int k0 = ks * 32;
        v8h a0 = *(const v8h*)&lA[arow * KPAD + k0 + abase0];
        v8h a1 = *(const v8h*)&lA[arow * KPAD + k0 + abase0 + 8];
        v16h a;
        #pragma unroll
        for (int j = 0; j < 8; ++j) { a[j] = a0[j]; a[j + 8] = a1[j]; }
        #pragma unroll
        for (int nt = 0; nt < 6; ++nt) {
            const v16h bf = *(const v16h*)&wpack[(nt * 16 + bn) * KPAD + k0 + bkoff];
            acc[nt] = __builtin_amdgcn_wmma_f32_16x16x32_f16(
                false, a, false, bf, (short)0, acc[nt], false, false);
        }
    }

    const int mrow_off = wave * 16 + ((lane >> 4) << 3);
    #pragma unroll
    for (int nt = 0; nt < 6; ++nt) {
        const int oc = nt * 16 + bn;
        const float bv = bias[oc];
        #pragma unroll
        for (int r = 0; r < 8; ++r) {
            int m  = mbase + mrow_off + r;
            int ow = m % OW; int tmp = m / OW; int oh = tmp % OH; int b = tmp / OH;
            y[((b * OCN + oc) * OH + oh) * OW + ow] = acc[nt][r] + bv;
        }
    }
}

// ---------------- fused LCN (mean+var) + ReLU + 2x2 maxpool -----------------
__global__ __launch_bounds__(256) void lcn_pool(
    const float* __restrict__ y, float* __restrict__ pre,
    float* __restrict__ out, int* __restrict__ idx)
{
    __shared__ float ty[24 * 24];   // y tile with 4-halo
    __shared__ float tc[20 * 20];   // centered tile with 2-halo
    __shared__ float tp[16 * 16];   // pre tile for pooling
    const int plane = blockIdx.z;
    const float* yp = y + (size_t)plane * PLANE;
    const int tx0 = blockIdx.x * 16, ty0 = blockIdx.y * 16;
    const int tid = threadIdx.x;

    for (int i = tid; i < 24 * 24; i += 256) {
        int r = i / 24, c = i % 24;
        int gy = ty0 + r - 4, gx = tx0 + c - 4;
        ty[i] = (gy >= 0 && gy < OH && gx >= 0 && gx < OW) ? yp[gy * OW + gx] : 0.f;
    }
    __syncthreads();

    // centered = y - box5(y); explicitly zero outside the image, because the
    // reference zero-pads centered^2 in the variance box conv.
    for (int i = tid; i < 20 * 20; i += 256) {
        int r = i / 20, c = i % 20;
        int gy = ty0 + r - 2, gx = tx0 + c - 2;
        float v = 0.f;
        if (gy >= 0 && gy < OH && gx >= 0 && gx < OW) {
            float s = 0.f;
            #pragma unroll
            for (int dy = 0; dy < 5; ++dy)
                #pragma unroll
                for (int dx = 0; dx < 5; ++dx) s += ty[(r + dy) * 24 + (c + dx)];
            v = ty[(r + 2) * 24 + (c + 2)] - s * (1.f / 25.f);
        }
        tc[i] = v;
    }
    __syncthreads();

    const int lx = tid & 15, ly = tid >> 4;
    float s = 0.f;
    #pragma unroll
    for (int dy = 0; dy < 5; ++dy)
        #pragma unroll
        for (int dx = 0; dx < 5; ++dx) { float v = tc[(ly + dy) * 20 + (lx + dx)]; s += v * v; }
    const float cv = tc[(ly + 2) * 20 + (lx + 2)];
    const float pv = fmaxf(cv * rsqrtf(s * (1.f / 25.f) + 1.f), 0.f);
    pre[(size_t)plane * PLANE + (ty0 + ly) * OW + (tx0 + lx)] = pv;
    tp[ly * 16 + lx] = pv;
    __syncthreads();

    if (tid < 64) {
        int py = tid >> 3, px = tid & 7;
        int h0 = py * 2, w0 = px * 2;
        float v0 = tp[h0 * 16 + w0],       v1 = tp[h0 * 16 + w0 + 1];
        float v2 = tp[(h0 + 1) * 16 + w0], v3 = tp[(h0 + 1) * 16 + w0 + 1];
        float m = v0; int loc = 0;          // first max wins (strict >)
        if (v1 > m) { m = v1; loc = 1; }
        if (v2 > m) { m = v2; loc = 2; }
        if (v3 > m) { m = v3; loc = 3; }
        int ph = (ty0 >> 1) + py, pw = (tx0 >> 1) + px;
        size_t o = (size_t)plane * 56 * 56 + ph * 56 + pw;
        out[o] = m;
        idx[o] = (ty0 + h0 + (loc >> 1)) * OW + (tx0 + w0 + (loc & 1));
    }
}

extern "C" void kernel_launch(void* const* d_in, const int* in_sizes, int n_in,
                              void* d_out, int out_size, void* d_ws, size_t ws_size,
                              hipStream_t stream) {
    (void)in_sizes; (void)n_in; (void)out_size; (void)ws_size;
    const float* x    = (const float*)d_in[0];   // [16,3,224,224]
    const float* w    = (const float*)d_in[1];   // [96,3,7,7]
    const float* bias = (const float*)d_in[2];   // [96]

    float* out = (float*)d_out;                  // [16,96,56,56]
    float* pre = out + POOLN;                    // [16,96,112,112]
    int*   idx = (int*)(pre + NPLANES * PLANE);  // [16,96,56,56] int32

    _Float16* wpack = (_Float16*)d_ws;                       // 96*160 f16
    float*    ybuf  = (float*)((char*)d_ws + 65536);         // 77 MB y buffer

    wrenorm_pack<<<1, 128, 0, stream>>>(w, wpack);
    // y (conv+bias) goes to workspace so the fused kernel's halo reads can
    // never race with its pre_pool writes (different buffers, RAW only
    // across kernel launches).
    conv_wmma<<<(BATCH * PLANE) / 64, 128, 0, stream>>>(x, wpack, bias, ybuf);
    lcn_pool<<<dim3(OW / 16, OH / 16, NPLANES), 256, 0, stream>>>(ybuf, pre, out, idx);
}
